// SplitLSTM_50345606644309
// MI455X (gfx1250) — matrix operand order repaired and checked by reading, not compiled, not run on hardware
//
#include <hip/hip_runtime.h>
#include <hip/hip_bf16.h>

#define NUM_CONST   29
#define NUM_VAR     3
#define NUM_HIDDEN  5
#define NUM_GROUPS  30490
#define SEQ_LEN     128
#define INPUT_SIZE  (NUM_CONST + NUM_VAR * NUM_GROUPS)   // 91499
#define HIDDEN_SIZE (NUM_HIDDEN * NUM_GROUPS)            // 152450
#define KPAD        32
#define GPB         8                                    // groups per block
#define RPK         (GPB * NUM_HIDDEN)                   // 40 rows per gate k
#define ROWS        (4 * RPK)                            // 160 gate-rows per block
#define ROW_TILES   (ROWS / 16)                          // 10
#define THREADS     256

// ---- LDS byte-offset map (all 16B aligned) ----
#define LDS_GI      0u                                   // 128*160 f32 = 81920
#define LDS_WBF     81920u                               // 160*32 bf16 = 10240
#define LDS_XCBF    92160u                               // 128*32 bf16 = 8192
#define LDS_WSTG    100352u                              // 4*1160 f32  = 18560
#define LDS_XCSTG   118912u                              // 128*29 f32  = 14848
#define LDS_XVSTG   133760u                              // 128*24 f32  = 12288
#define LDS_WHH     146048u                              // 4*200 f32   = 3200
#define LDS_WV      149248u                              // 4*120 f32   = 1920
#define LDS_BIH     151168u                              // 160 f32     = 640
#define LDS_BHH     151808u                              // 160 f32     = 640
#define LDS_TOTAL   152448u

typedef __bf16 v16bf __attribute__((ext_vector_type(16)));
typedef __bf16 bf8   __attribute__((ext_vector_type(8)));
typedef float  v8f   __attribute__((ext_vector_type(8)));
typedef unsigned int u32x4 __attribute__((ext_vector_type(4)));
typedef int          i32x4 __attribute__((ext_vector_type(4)));
typedef int          i32x8 __attribute__((ext_vector_type(8)));

union Frag { v16bf v; bf8 h[2]; };

__device__ __forceinline__ float sigmoid_(float x) {
    return 1.0f / (1.0f + __expf(-x));
}
__device__ __forceinline__ float tanh_(float x) {
    float t = __expf(-2.0f * fabsf(x));
    float r = (1.0f - t) / (1.0f + t);
    return (x >= 0.0f) ? r : -r;
}

// Issue one 2D TDM tile load (f32 elements) global -> LDS.
// D# group0: count=1 | lds_addr | global_addr(57b) | type=2
// D# group1: data_size=4B, tensor_dim0/1, tile_dim0/1, tensor_dim0_stride
__device__ __forceinline__ void tdm_load_2d_f32(unsigned lds_byte_addr,
                                                const void* gptr,
                                                unsigned tile_d0, unsigned tile_d1,
                                                unsigned tensor_d0, unsigned tensor_d1,
                                                unsigned long long stride0_elems) {
    unsigned long long ga = (unsigned long long)(uintptr_t)gptr;
    u32x4 g0;
    g0[0] = 1u;                                            // count=1, user descriptor
    g0[1] = lds_byte_addr;                                 // LDS dest (bytes)
    g0[2] = (unsigned)ga;                                  // global_addr[31:0]
    g0[3] = (unsigned)((ga >> 32) & 0x1FFFFFFu) | 0x80000000u; // addr[56:32] | type=2
    i32x8 g1;
    g1[0] = (int)(2u << 16);                               // data_size = 4 bytes
    g1[1] = (int)((tensor_d0 & 0xFFFFu) << 16);            // tensor_dim0[15:0]
    g1[2] = (int)(((tensor_d0 >> 16) & 0xFFFFu) |
                  ((tensor_d1 & 0xFFFFu) << 16));          // tensor_dim0[31:16] | tensor_dim1[15:0]
    g1[3] = (int)(((tensor_d1 >> 16) & 0xFFFFu) |
                  ((tile_d0 & 0xFFFFu) << 16));            // tensor_dim1[31:16] | tile_dim0
    g1[4] = (int)(tile_d1 & 0xFFFFu);                      // tile_dim1 (tile_dim2 = 0)
    g1[5] = (int)(stride0_elems & 0xFFFFFFFFull);          // tensor_dim0_stride[31:0]
    g1[6] = (int)((stride0_elems >> 32) & 0xFFFFull);      // stride[47:32] (dim1_stride unused)
    g1[7] = 0;
    i32x4 z4 = {0, 0, 0, 0};
    i32x8 z8 = {0, 0, 0, 0, 0, 0, 0, 0};
    __builtin_amdgcn_tensor_load_to_lds(g0, g1, z4, z4, z8, 0);
}

__global__ __launch_bounds__(THREADS)
void splitlstm_fused(const float* __restrict__ x,     // (128, 91499)
                     const float* __restrict__ Wc,    // (4, G, 5, 29)
                     const float* __restrict__ Wv,    // (4, G, 5, 3)
                     const float* __restrict__ Whh,   // (4, G, 5, 5)
                     const float* __restrict__ bih,   // (4, G, 5)
                     const float* __restrict__ bhh,   // (4, G, 5)
                     float* __restrict__ out)         // (128, 152450)
{
    extern __shared__ char smem_raw[];
    const unsigned lds_base = (unsigned)(uintptr_t)(void*)smem_raw;

    float*  gi      = (float*)(smem_raw + LDS_GI);
    __bf16* w_bf    = (__bf16*)(smem_raw + LDS_WBF);
    __bf16* xc_bf   = (__bf16*)(smem_raw + LDS_XCBF);
    const float* wstg  = (const float*)(smem_raw + LDS_WSTG);
    const float* xcstg = (const float*)(smem_raw + LDS_XCSTG);
    const float* xvstg = (const float*)(smem_raw + LDS_XVSTG);
    const float* whh_s = (const float*)(smem_raw + LDS_WHH);
    const float* wv_s  = (const float*)(smem_raw + LDS_WV);
    const float* bih_s = (const float*)(smem_raw + LDS_BIH);
    const float* bhh_s = (const float*)(smem_raw + LDS_BHH);

    const int tid  = threadIdx.x;
    const int g0   = blockIdx.x * GPB;
    const int wave = tid >> 5;
    const int lane = tid & 31;
    const unsigned remE = (unsigned)(NUM_GROUPS - g0);   // groups remaining (>=1)

    // ---------------- Phase 0: TDM DMA of all block operands into LDS ----------------
    // One descriptor per wave; OOB reads past tensor_dim0 return zero (handles last block).
    if (wave == 0) {          // W_const block: 4 x 1160 f32, row stride G*145
        unsigned td0 = remE * 145u < 1160u ? remE * 145u : 1160u;
        tdm_load_2d_f32(lds_base + LDS_WSTG, Wc + (size_t)g0 * 145, 1160, 4, td0, 4,
                        (unsigned long long)NUM_GROUPS * 145ull);
    } else if (wave == 1) {   // const input: 128 x 29 f32, row stride 91499
        tdm_load_2d_f32(lds_base + LDS_XCSTG, x, 29, SEQ_LEN, 29, SEQ_LEN,
                        (unsigned long long)INPUT_SIZE);
    } else if (wave == 2) {   // var input slice: 128 x 24 f32
        unsigned td0 = remE * 3u < 24u ? remE * 3u : 24u;
        tdm_load_2d_f32(lds_base + LDS_XVSTG, x + NUM_CONST + (size_t)g0 * 3, 24, SEQ_LEN,
                        td0, SEQ_LEN, (unsigned long long)INPUT_SIZE);
    } else if (wave == 3) {   // W_hh block: 4 x 200 f32
        unsigned td0 = remE * 25u < 200u ? remE * 25u : 200u;
        tdm_load_2d_f32(lds_base + LDS_WHH, Whh + (size_t)g0 * 25, 200, 4, td0, 4,
                        (unsigned long long)NUM_GROUPS * 25ull);
    } else if (wave == 4) {   // W_var block: 4 x 120 f32
        unsigned td0 = remE * 15u < 120u ? remE * 15u : 120u;
        tdm_load_2d_f32(lds_base + LDS_WV, Wv + (size_t)g0 * 15, 120, 4, td0, 4,
                        (unsigned long long)NUM_GROUPS * 15ull);
    } else if (wave == 5) {   // b_ih block: 4 x 40 f32
        unsigned td0 = remE * 5u < 40u ? remE * 5u : 40u;
        tdm_load_2d_f32(lds_base + LDS_BIH, bih + (size_t)g0 * 5, 40, 4, td0, 4,
                        (unsigned long long)NUM_GROUPS * 5ull);
    } else if (wave == 6) {   // b_hh block: 4 x 40 f32
        unsigned td0 = remE * 5u < 40u ? remE * 5u : 40u;
        tdm_load_2d_f32(lds_base + LDS_BHH, bhh + (size_t)g0 * 5, 40, 4, td0, 4,
                        (unsigned long long)NUM_GROUPS * 5ull);
    }
    __builtin_amdgcn_s_wait_tensorcnt(0);
    __syncthreads();

    // ---------------- Phase A0: LDS f32 -> padded bf16 operand tiles ----------------
    for (int idx = tid; idx < ROWS * KPAD; idx += THREADS) {   // 20 iters
        int r = idx >> 5, c = idx & 31;
        int k = r / RPK, rem = r % RPK;                        // rem = gl*5+h
        float v = (c < NUM_CONST) ? wstg[k * 1160 + rem * NUM_CONST + c] : 0.0f;
        w_bf[idx] = (__bf16)v;
    }
    for (int idx = tid; idx < SEQ_LEN * KPAD; idx += THREADS) { // 16 iters
        int s = idx >> 5, c = idx & 31;
        float v = (c < NUM_CONST) ? xcstg[s * NUM_CONST + c] : 0.0f;
        xc_bf[idx] = (__bf16)v;
    }
    __syncthreads();

    // ---------------- Phase A1: WMMA GEMM  gi = W_const @ xc^T ----------------
    // All 10 accumulator tiles live in registers; A fragments are double-buffered so
    // the ds_load of tile rt+1 overlaps the WMMA of tile rt (no stores in the loop).
    {
        const int scol = wave * 16 + (lane & 15);
        const int kb_b = (lane < 16) ? 0 : 8;
        Frag B;
        B.h[0] = *(const bf8*)&xc_bf[scol * KPAD + kb_b];
        B.h[1] = *(const bf8*)&xc_bf[scol * KPAD + kb_b + 16];

        const int kb_a  = (lane < 16) ? 0 : 8;
        const int arow  = lane & 15;
        v8f  accs[ROW_TILES];
        Frag A[2];

        A[0].h[0] = *(const bf8*)&w_bf[(0 * 16 + arow) * KPAD + kb_a];
        A[0].h[1] = *(const bf8*)&w_bf[(0 * 16 + arow) * KPAD + kb_a + 16];
        #pragma unroll
        for (int rt = 0; rt < ROW_TILES; ++rt) {
            if (rt + 1 < ROW_TILES) {
                int row = (rt + 1) * 16 + arow;
                A[(rt + 1) & 1].h[0] = *(const bf8*)&w_bf[row * KPAD + kb_a];
                A[(rt + 1) & 1].h[1] = *(const bf8*)&w_bf[row * KPAD + kb_a + 16];
            }
            v8f c0 = {};
            accs[rt] = __builtin_amdgcn_wmma_f32_16x16x32_bf16(
                           false, A[rt & 1].v, false, B.v, (short)0, c0, false, false);
        }

        const int N = wave * 16 + (lane & 15);
        const int mh = (lane >> 4) << 3;
        #pragma unroll
        for (int rt = 0; rt < ROW_TILES; ++rt) {
            int M = rt * 16 + mh;
            float4 lo = {accs[rt][0], accs[rt][1], accs[rt][2], accs[rt][3]};
            float4 hi = {accs[rt][4], accs[rt][5], accs[rt][6], accs[rt][7]};
            *(float4*)&gi[N * ROWS + M]     = lo;
            *(float4*)&gi[N * ROWS + M + 4] = hi;
        }
    }
    __syncthreads();

    // ---------------- Phase A2: add per-group var term + biases (all LDS) ----------------
    for (int idx = tid; idx < SEQ_LEN * GPB; idx += THREADS) {  // 4 iters
        int s = idx >> 3, gl = idx & 7;
        float x0 = xvstg[s * 24 + gl * 3 + 0];
        float x1 = xvstg[s * 24 + gl * 3 + 1];
        float x2 = xvstg[s * 24 + gl * 3 + 2];
        #pragma unroll
        for (int k = 0; k < 4; ++k) {
            #pragma unroll
            for (int h = 0; h < NUM_HIDDEN; ++h) {
                int r = k * RPK + gl * NUM_HIDDEN + h;
                gi[s * ROWS + r] += wv_s[r * 3] * x0 + wv_s[r * 3 + 1] * x1
                                  + wv_s[r * 3 + 2] * x2 + bih_s[r] + bhh_s[r];
            }
        }
    }
    __syncthreads();

    // ---------------- Phase B: 128-step recurrence (2 waves, shfl only) ----------------
    if (wave < 2) {
        const int lgrp = lane >> 3;            // 4 groups per wave, 8 lanes per group
        const int h    = lane & 7;             // active h < 5
        const int gl   = wave * 4 + lgrp;
        const int g    = g0 + gl;
        const bool act = (h < NUM_HIDDEN) && (g < NUM_GROUPS);
        const int  hc  = (h < NUM_HIDDEN) ? h : 0;

        float whh[4][NUM_HIDDEN];
        #pragma unroll
        for (int k = 0; k < 4; ++k)
            #pragma unroll
            for (int j = 0; j < NUM_HIDDEN; ++j)
                whh[k][j] = whh_s[k * 200 + (gl * NUM_HIDDEN + hc) * NUM_HIDDEN + j];

        const int rbase = gl * NUM_HIDDEN + hc;
        const int lb    = lane & ~7;
        float hst = 0.0f, cst = 0.0f;

        for (int s = 0; s < SEQ_LEN; ++s) {
            float hj[NUM_HIDDEN];
            #pragma unroll
            for (int j = 0; j < NUM_HIDDEN; ++j)
                hj[j] = __shfl(hst, lb + j, 32);

            float gate[4];
            #pragma unroll
            for (int k = 0; k < 4; ++k) {
                float a = gi[s * ROWS + k * RPK + rbase];
                #pragma unroll
                for (int j = 0; j < NUM_HIDDEN; ++j)
                    a += whh[k][j] * hj[j];
                gate[k] = a;
            }
            float ig = sigmoid_(gate[0]);
            float fg = sigmoid_(gate[1]);
            float gg = tanh_(gate[2]);
            float og = sigmoid_(gate[3]);
            cst = fg * cst + ig * gg;
            float hn = og * tanh_(cst);
            hst = hn;
            if (act)
                out[(size_t)s * HIDDEN_SIZE + (size_t)g * NUM_HIDDEN + h] = hn;
        }
    }
}

extern "C" void kernel_launch(void* const* d_in, const int* in_sizes, int n_in,
                              void* d_out, int out_size, void* d_ws, size_t ws_size,
                              hipStream_t stream) {
    (void)in_sizes; (void)n_in; (void)d_ws; (void)ws_size; (void)out_size;
    const float* x   = (const float*)d_in[0];
    const float* Wc  = (const float*)d_in[1];
    const float* Wv  = (const float*)d_in[2];
    const float* Whh = (const float*)d_in[3];
    const float* bih = (const float*)d_in[4];
    const float* bhh = (const float*)d_in[5];
    float* out = (float*)d_out;

    const int blocks = (NUM_GROUPS + GPB - 1) / GPB;   // 3812
    splitlstm_fused<<<blocks, THREADS, LDS_TOTAL, stream>>>(x, Wc, Wv, Whh, bih, bhh, out);
}
